// PrimalAttention_10582799417408
// MI455X (gfx1250) — compile-verified, matrix operand order checked by reading, not run on hardware
//
#include <hip/hip_runtime.h>
#include <hip/hip_bf16.h>
#include <stdint.h>

// ---------------- types / helpers ----------------
typedef __attribute__((ext_vector_type(16))) __bf16       bf16x16;
typedef __attribute__((ext_vector_type(8)))  float        f32x8;
typedef __attribute__((ext_vector_type(4)))  unsigned int u32x4;
typedef __attribute__((ext_vector_type(4)))  int          v4i;

union FragBF16 { u32x4 u4[2]; bf16x16 v; };

__device__ __forceinline__ unsigned short f32_to_bf16(float f) {
  unsigned u = __float_as_uint(f);
  u = (u + 0x7FFFu + ((u >> 16) & 1u)) >> 16;   // round-to-nearest-even
  return (unsigned short)u;
}

__device__ __forceinline__ f32x8 wmma_bf16(bf16x16 a, bf16x16 b, f32x8 c) {
  return __builtin_amdgcn_wmma_f32_16x16x32_bf16(false, a, false, b,
                                                 (short)0, c, false, false);
}

// ---- CDNA5 async global->LDS copy (ASYNCcnt), with safe fallback ----
#if defined(__has_builtin)
#  if __has_builtin(__builtin_amdgcn_global_load_async_to_lds_b128)
#    define HAS_ASYNC_LDS 1
#  endif
#  if __has_builtin(__builtin_amdgcn_s_wait_asynccnt)
#    define HAS_WAIT_ASYNC 1
#  endif
#endif

__device__ __forceinline__ void g2l_b128(const unsigned short* g, unsigned short* l) {
#if defined(HAS_ASYNC_LDS)
  // param0: v4i __device__* (AS1 global src), param1: v4i __shared__* (AS3 lds dst)
  __builtin_amdgcn_global_load_async_to_lds_b128(
      (__attribute__((address_space(1))) v4i*)(g),
      (__attribute__((address_space(3))) v4i*)(l),
      0, 0);
#else
  *(u32x4*)l = *(const u32x4*)g;
#endif
}

__device__ __forceinline__ void wait_async() {
#if defined(HAS_ASYNC_LDS)
#  if defined(HAS_WAIT_ASYNC)
  __builtin_amdgcn_s_wait_asynccnt(0);
#  else
  asm volatile("s_wait_asynccnt 0" ::: "memory");
#  endif
#endif
}

// ---------------- kernel 0: fp32 -> bf16 convert ----------------
__global__ void cvt_f32_bf16(const float* __restrict__ src,
                             unsigned short* __restrict__ dst, int n) {
  int i = (blockIdx.x * blockDim.x + threadIdx.x) * 4;
  if (i + 3 < n) {
    float4 f = *(const float4*)(src + i);
    unsigned lo = (unsigned)f32_to_bf16(f.x) | ((unsigned)f32_to_bf16(f.y) << 16);
    unsigned hi = (unsigned)f32_to_bf16(f.z) | ((unsigned)f32_to_bf16(f.w) << 16);
    uint2 o; o.x = lo; o.y = hi;
    *(uint2*)(dst + i) = o;
  }
}

// ---------------- staged WMMA GEMM: C[M,N] = A[M,K] * W[N,K]^T ---------------
// block = 256 threads (8 waves), block tile 128x128, wave tile 32x64, K-step 32
// A,W row-major bf16. LDS double-buffered, filled with async global->LDS DMA.
// MODE 1: fp32 out + bias        (projection GEMM)
// MODE 2: bf16 out, l2-normalize each 64-aligned column chunk per row (qk GEMM)
constexpr int LDSTR = 40;  // padded LDS row stride (elements): 80B -> conflict-free

template <int MODE>
__global__ void gemm_bf16_staged(const unsigned short* __restrict__ A,
                                 const unsigned short* __restrict__ W,
                                 const float* __restrict__ bias,
                                 void* __restrict__ outp,
                                 int M, int N, int K) {
  __shared__ __align__(16) unsigned short As[2][128 * LDSTR];
  __shared__ __align__(16) unsigned short Bs[2][128 * LDSTR];

  const int tid  = threadIdx.x;
  const int wave = tid >> 5, lane = tid & 31;
  const int lrow = lane & 15, kh = lane >> 4;
  const int bm = blockIdx.x * 128, bn = blockIdx.y * 128;
  const int wmL = (wave & 3) * 32;   // wave tile origin inside block tile
  const int wnL = (wave >> 2) * 64;

  // staging: 512 16B-chunks per tile (128 rows x 4 chunks); thread does 2 per tile
  const int c0 = tid, c1 = tid + 256;
  const int ar0 = c0 >> 2, aq0 = (c0 & 3) * 8;
  const int ar1 = c1 >> 2, aq1 = (c1 & 3) * 8;

  auto stage = [&](int buf, int k0) {
    g2l_b128(A + (size_t)(bm + ar0) * K + k0 + aq0, &As[buf][ar0 * LDSTR + aq0]);
    g2l_b128(A + (size_t)(bm + ar1) * K + k0 + aq1, &As[buf][ar1 * LDSTR + aq1]);
    g2l_b128(W + (size_t)(bn + ar0) * K + k0 + aq0, &Bs[buf][ar0 * LDSTR + aq0]);
    g2l_b128(W + (size_t)(bn + ar1) * K + k0 + aq1, &Bs[buf][ar1 * LDSTR + aq1]);
  };

  const f32x8 zero = {0.f,0.f,0.f,0.f,0.f,0.f,0.f,0.f};
  f32x8 acc[2][4];
  for (int i = 0; i < 2; ++i)
    for (int j = 0; j < 4; ++j) acc[i][j] = zero;

  const int ktiles = K / 32;
  stage(0, 0);

  for (int kt = 0; kt < ktiles; ++kt) {
    const int cur = kt & 1;
    wait_async();          // my DMA into 'cur' (and previous) complete
    __syncthreads();       // everyone's DMA complete / reads of 'cur^1' done
    if (kt + 1 < ktiles) stage(cur ^ 1, (kt + 1) * 32);

    const unsigned short* As_c = &As[cur][0];
    const unsigned short* Bs_c = &Bs[cur][0];
    FragBF16 a[2], b[4];
#pragma unroll
    for (int i = 0; i < 2; ++i) {
      const unsigned short* p = As_c + (wmL + i * 16 + lrow) * LDSTR + kh * 8;
      a[i].u4[0] = *(const u32x4*)p;
      a[i].u4[1] = *(const u32x4*)(p + 16);
    }
#pragma unroll
    for (int j = 0; j < 4; ++j) {
      const unsigned short* p = Bs_c + (wnL + j * 16 + lrow) * LDSTR + kh * 8;
      b[j].u4[0] = *(const u32x4*)p;
      b[j].u4[1] = *(const u32x4*)(p + 16);
    }
#pragma unroll
    for (int i = 0; i < 2; ++i)
#pragma unroll
      for (int j = 0; j < 4; ++j)
        acc[i][j] = wmma_bf16(a[i].v, b[j].v, acc[i][j]);
  }

  // ---- epilogue ----
  // D layout: lane = col + 16*(row>=8 of tile); vgpr r holds row r + 8*(lane>=16)
  const int colL = lane & 15;
  const int rsub = kh * 8;

  if (MODE == 1) {
    float* out = (float*)outp;
#pragma unroll
    for (int i = 0; i < 2; ++i)
#pragma unroll
      for (int j = 0; j < 4; ++j) {
        int col = bn + wnL + j * 16 + colL;
        float bv = bias[col];
#pragma unroll
        for (int r = 0; r < 8; ++r) {
          int row = bm + wmL + i * 16 + rsub + r;
          out[(size_t)row * N + col] = acc[i][j][r] + bv;
        }
      }
  } else {
    // l2-normalize the wave's 64-column chunk per output row, emit bf16.
    // For a fixed row, its 64 values live in one 16-lane half across j=0..3.
    unsigned short* out = (unsigned short*)outp;
#pragma unroll
    for (int i = 0; i < 2; ++i) {
#pragma unroll
      for (int r = 0; r < 8; ++r) {
        float s = 0.f;
#pragma unroll
        for (int j = 0; j < 4; ++j) s += acc[i][j][r] * acc[i][j][r];
        s += __shfl_xor(s, 1, 32);
        s += __shfl_xor(s, 2, 32);
        s += __shfl_xor(s, 4, 32);
        s += __shfl_xor(s, 8, 32);        // masks < 16: stays in the row's half
        float inv = 1.0f / fmaxf(sqrtf(s), 1e-12f);
        int row = bm + wmL + i * 16 + rsub + r;
#pragma unroll
        for (int j = 0; j < 4; ++j) {
          int col = bn + wnL + j * 16 + colL;
          out[(size_t)row * N + col] = f32_to_bf16(acc[i][j][r] * inv);
        }
      }
    }
  }
}

// ---------------- kernel 3: fold low-rank path into per-(b,h) 64x128 matrix --
__global__ void build_mw(const float* __restrict__ x,
                         const float* __restrict__ we,
                         const float* __restrict__ wr,
                         const float* __restrict__ cw_w,
                         unsigned short* __restrict__ Mw) {
  constexpr int R = 200, LR = 20, HD = 64, N = 2048, C = 1024;
  __shared__ float xh[HD][R];
  __shared__ float web[HD][LR];
  __shared__ float wrb[HD][LR];
  const int bh = blockIdx.x, b = bh >> 4, h = bh & 15;
  const int tid = threadIdx.x;

  // idx[r] = trunc(linspace(0, N-1, R)) == (r*2047)/199 exactly
  for (int t = tid; t < HD * R; t += blockDim.x) {
    int d = t / R, r = t - d * R;
    int idx = (r * (N - 1)) / (R - 1);
    xh[d][r] = x[((size_t)b * N + idx) * C + h * HD + d];
  }
  __syncthreads();

  for (int t = tid; t < 2 * HD * LR; t += blockDim.x) {
    int sel = (t >= HD * LR);
    int t2  = sel ? t - HD * LR : t;
    int d = t2 / LR, e = t2 - d * LR;
    const float* w = (sel ? wr : we) + ((size_t)h * R) * LR + e;
    float s = 0.f;
    for (int r = 0; r < R; ++r) s += xh[d][r] * w[(size_t)r * LR];
    float* dst = sel ? &wrb[0][0] : &web[0][0];
    dst[d * LR + e] = s;
  }
  __syncthreads();

  for (int t = tid; t < HD * 128; t += blockDim.x) {
    int dout = t >> 7;
    int kidx = t & 127;
    float s = 0.f;
    if (kidx < HD) {
      const float* cw = cw_w + dout * 2 * LR;
      for (int e = 0; e < LR; ++e) s += cw[e] * web[kidx][e];
    } else {
      const float* cw = cw_w + dout * 2 * LR + LR;
      for (int e = 0; e < LR; ++e) s += cw[e] * wrb[kidx - HD][e];
    }
    Mw[(size_t)bh * HD * 128 + t] = f32_to_bf16(s);
  }
}

// ---------------- kernel 4: per-(b,h) attn = [q|k][2048x128] @ Mw^T + cw_b ---
__global__ void head_attn_gemm(const unsigned short* __restrict__ qkn,
                               const unsigned short* __restrict__ Mw,
                               const float* __restrict__ cw_b,
                               unsigned short* __restrict__ attn) {
  constexpr int N = 2048, C = 1024, HD = 64;
  const int bh = blockIdx.x >> 4, mblk = blockIdx.x & 15;
  const int b = bh >> 4, h = bh & 15;
  const int wave = threadIdx.x >> 5, lane = threadIdx.x & 31;
  const int row0 = mblk * 128 + wave * 16;
  const int lrow = lane & 15, kh = lane >> 4;

  const unsigned short* abase =
      qkn + ((size_t)(b * N + row0 + lrow)) * (2 * C) + h * HD;
  const unsigned short* wbase =
      Mw + (size_t)bh * HD * 128 + (size_t)lrow * 128 + kh * 8;

  const f32x8 zero = {0.f,0.f,0.f,0.f,0.f,0.f,0.f,0.f};
  f32x8 acc[4] = {zero, zero, zero, zero};

#pragma unroll
  for (int k0 = 0; k0 < 128; k0 += 32) {
    int kk0 = k0 + kh * 8;
    int kk1 = kk0 + 16;
    int e0 = (kk0 < 64) ? kk0 : kk0 + (C - HD);   // q chunk | k chunk (+960)
    int e1 = (kk1 < 64) ? kk1 : kk1 + (C - HD);
    FragBF16 a;
    a.u4[0] = *(const u32x4*)(abase + e0);
    a.u4[1] = *(const u32x4*)(abase + e1);
#pragma unroll
    for (int j = 0; j < 4; ++j) {
      FragBF16 w;
      const unsigned short* p = wbase + j * 16 * 128 + k0;
      w.u4[0] = *(const u32x4*)p;
      w.u4[1] = *(const u32x4*)(p + 16);
      acc[j] = wmma_bf16(a.v, w.v, acc[j]);
    }
  }

  const int colL = lane & 15, rsub = kh * 8;
#pragma unroll
  for (int j = 0; j < 4; ++j) {
    int d = j * 16 + colL;
    float bv = cw_b[d];
#pragma unroll
    for (int r = 0; r < 8; ++r) {
      int n = row0 + rsub + r;
      attn[((size_t)(b * N + n)) * C + h * HD + d] = f32_to_bf16(acc[j][r] + bv);
    }
  }
}

// ---------------- launcher ----------------
extern "C" void kernel_launch(void* const* d_in, const int* in_sizes, int n_in,
                              void* d_out, int out_size, void* d_ws, size_t ws_size,
                              hipStream_t stream) {
  const float* x      = (const float*)d_in[0];
  const float* qk_w   = (const float*)d_in[1];
  const float* proj_w = (const float*)d_in[2];
  const float* proj_b = (const float*)d_in[3];
  const float* we     = (const float*)d_in[4];
  const float* wr     = (const float*)d_in[5];
  const float* cw_w   = (const float*)d_in[6];
  const float* cw_b   = (const float*)d_in[7];
  float* out = (float*)d_out;

  constexpr size_t B = 4, N = 2048, C = 1024, H = 16;
  constexpr size_t M = B * N;                 // 8192 rows

  char* ws = (char*)d_ws;
  size_t off = 0;
  auto alloc = [&](size_t bytes) -> void* {
    void* p = (void*)(ws + off);
    off += (bytes + 255) & ~(size_t)255;
    return p;
  };
  unsigned short* x_bf   = (unsigned short*)alloc(M * C * 2);            // 16 MB
  unsigned short* qkw_bf = (unsigned short*)alloc(2 * C * C * 2);        //  4 MB
  unsigned short* pw_bf  = (unsigned short*)alloc(C * C * 2);            //  2 MB
  unsigned short* qkn    = (unsigned short*)alloc(M * 2 * C * 2);        // 32 MB
  unsigned short* Mw     = (unsigned short*)alloc(B * H * 64 * 128 * 2); //  1 MB
  unsigned short* attn   = (unsigned short*)alloc(M * C * 2);            // 16 MB

  // 1) fp32 -> bf16 converts
  cvt_f32_bf16<<<(int)(M * C / 1024), 256, 0, stream>>>(x, x_bf, (int)(M * C));
  cvt_f32_bf16<<<(int)(2 * C * C / 1024), 256, 0, stream>>>(qk_w, qkw_bf, (int)(2 * C * C));
  cvt_f32_bf16<<<(int)(C * C / 1024), 256, 0, stream>>>(proj_w, pw_bf, (int)(C * C));

  // 2) qk = x @ qk_w^T, fused per-head-row l2-norm -> bf16 qkn  [8192 x 2048]
  gemm_bf16_staged<2><<<dim3(M / 128, 2 * C / 128), 256, 0, stream>>>(
      x_bf, qkw_bf, nullptr, qkn, (int)M, (int)(2 * C), (int)C);

  // 3) fold low-rank weights into per-(b,h) 64x128 Mw
  build_mw<<<(int)(B * H), 256, 0, stream>>>(x, we, wr, cw_w, Mw);

  // 4) attn[b,h] = [q|k] @ Mw^T + cw_b  -> bf16 in [B,N,C] layout
  head_attn_gemm<<<(int)(B * H * 16), 256, 0, stream>>>(qkn, Mw, cw_b, attn);

  // 5) out = attn @ proj_w^T + proj_b   [8192 x 1024]
  gemm_bf16_staged<1><<<dim3(M / 128, C / 128), 256, 0, stream>>>(
      attn, pw_bf, proj_b, out, (int)M, (int)C, (int)C);
}
